// ConvStackedTemporalGCN_31722628448360
// MI455X (gfx1250) — matrix operand order repaired and checked by reading, not compiled, not run on hardware
//
#include <hip/hip_runtime.h>

#define N_NODES 10000
#define N_EDGES 160000
#define FEAT    4
#define PER     12
#define CH      512
#define HIDN    256
#define OUTD    12

typedef __bf16 bf16_t;
typedef __attribute__((ext_vector_type(16))) __bf16    v16bf;
typedef __attribute__((ext_vector_type(8)))  __bf16    v8bf;
typedef __attribute__((ext_vector_type(8)))  float     v8f;
typedef __attribute__((ext_vector_type(4)))  float     v4f;
typedef __attribute__((ext_vector_type(4)))  unsigned  v4u;
typedef __attribute__((ext_vector_type(8)))  unsigned  v8u;

static __device__ __forceinline__ bf16_t to_bf16(float f) {
  unsigned u = __float_as_uint(f);
  u += 0x7FFFu + ((u >> 16) & 1u);            // round-to-nearest-even
  unsigned short s = (unsigned short)(u >> 16);
  return __builtin_bit_cast(bf16_t, s);
}

// ---------------------------------------------------------------------------
// Tensor Data Mover: DMA one [rows x 32] bf16 tile (row stride = strideK elems)
// from global into LDS. 2D descriptor per CDNA5 ISA 8.3/8.4:
//   group0: [1:0]=count=1, [63:32]=lds_addr, [120:64]=global_addr, [127:126]=2
//   group1: data_size=1(2B), tensor_dim0=k_rem, tensor_dim1=rows_rem (OOB rows
//           read as zero -> free M-edge handling), tile_dim0=32, tile_dim1=128,
//           tensor_dim0_stride=strideK
// ---------------------------------------------------------------------------
static __device__ __forceinline__ void tdm_load_tile(unsigned lds_off,
                                                     const bf16_t* gsrc,
                                                     unsigned rows_rem,
                                                     unsigned k_rem,
                                                     unsigned strideK) {
  unsigned long long ga = (unsigned long long)(uintptr_t)gsrc;
  v4u g0;
  g0[0] = 1u;                                  // count=1, no gather
  g0[1] = lds_off;                             // LDS byte address
  g0[2] = (unsigned)ga;                        // global addr lo
  g0[3] = (unsigned)(ga >> 32) | 0x80000000u;  // addr hi | type=2 ("image")
  v8u g1;
  g1[0] = 0x00010000u;                         // wg_mask=0, data_size=1 (2B)
  g1[1] = (k_rem & 0xFFFFu) << 16;             // tensor_dim0[15:0]
  g1[2] = (k_rem >> 16) | ((rows_rem & 0xFFFFu) << 16);  // td0[31:16] | td1[15:0]
  g1[3] = (rows_rem >> 16) | (32u << 16);      // td1[31:16] | tile_dim0=32
  g1[4] = 128u;                                // tile_dim1=128, tile_dim2=0
  g1[5] = strideK;                             // tensor_dim0_stride[31:0]
  g1[6] = 0u;                                  // stride hi | dim1_stride lo
  g1[7] = 0u;
  asm volatile("tensor_load_to_lds %0, %1" :: "s"(g0), "s"(g1) : "memory");
}

// ---------------------------------------------------------------------------
// WMMA GEMM: D[M,Nc] = A[M,K](bf16) @ Bt[Nc,K](bf16, pre-transposed), f32 out.
// 256 thr = 8 waves; tile 128x128x32; wave tile 64x32 (8 x 16x16 accums).
// A/B tiles staged by TDM, double-buffered on TENSORcnt.
// ---------------------------------------------------------------------------
#define BM 128
#define BN 128
#define BK 32

__global__ __launch_bounds__(256)
void k_gemm_wmma(const bf16_t* __restrict__ A, const bf16_t* __restrict__ Bt,
                 float* __restrict__ D, int M, int K, int Nc)
{
  __shared__ bf16_t Als[2][BM * BK];   // [m][k]
  __shared__ bf16_t Bls[2][BN * BK];   // [n][k]

  const int tid  = threadIdx.x;
  const int lane = tid & 31;
  const int wave = tid >> 5;
  const int wm   = (wave >> 2) * 64;
  const int wn   = (wave & 3) * 32;
  const int bm   = blockIdx.y * BM;
  const int bn   = blockIdx.x * BN;
  const unsigned rows_rem = (unsigned)(M - bm);

  v8f acc[4][2];
  for (int a = 0; a < 4; ++a)
    for (int b = 0; b < 2; ++b)
      for (int i = 0; i < 8; ++i) acc[a][b][i] = 0.0f;

  // fragment lane mapping per CDNA5 ISA (16-bit A 16x32 / B 32x16)
  const int frow  = lane & 15;
  const int khalf = (lane >> 4) << 3;
  const int klo   = (lane >> 4) << 4;

  if (wave == 0) {
    tdm_load_tile((unsigned)(uintptr_t)&Als[0][0], A + (size_t)bm * K,
                  rows_rem, (unsigned)K, (unsigned)K);
    tdm_load_tile((unsigned)(uintptr_t)&Bls[0][0], Bt + (size_t)bn * K,
                  128u, (unsigned)K, (unsigned)K);
  }

  for (int kb = 0; kb < K; kb += BK) {
    const int cur = (kb / BK) & 1;
    if (kb + BK < K) {
      if (wave == 0) {
        tdm_load_tile((unsigned)(uintptr_t)&Als[cur ^ 1][0],
                      A + (size_t)bm * K + kb + BK,
                      rows_rem, (unsigned)(K - kb - BK), (unsigned)K);
        tdm_load_tile((unsigned)(uintptr_t)&Bls[cur ^ 1][0],
                      Bt + (size_t)bn * K + kb + BK,
                      128u, (unsigned)(K - kb - BK), (unsigned)K);
      }
      __builtin_amdgcn_s_wait_tensorcnt(2);   // 2 newest outstanding: current done
    } else {
      __builtin_amdgcn_s_wait_tensorcnt(0);
    }
    __syncthreads();

    union FragU { v16bf v; v8bf h[2]; };
    FragU fa[4], fb[2];
    for (int mt = 0; mt < 4; ++mt) {
      const bf16_t* ap = &Als[cur][(wm + mt * 16 + frow) * BK];
      fa[mt].h[0] = *(const v8bf*)(ap + khalf);
      fa[mt].h[1] = *(const v8bf*)(ap + 16 + khalf);
    }
    for (int nt = 0; nt < 2; ++nt) {
      const bf16_t* bp = &Bls[cur][(wn + nt * 16 + frow) * BK + khalf];
      fb[nt].h[0] = *(const v8bf*)(bp);
      fb[nt].h[1] = *(const v8bf*)(bp + 16);
    }
    for (int mt = 0; mt < 4; ++mt)
      for (int nt = 0; nt < 2; ++nt)
        acc[mt][nt] = __builtin_amdgcn_wmma_f32_16x16x32_bf16(
            false, fa[mt].v, false, fb[nt].v, (short)0, acc[mt][nt], false, false);
    __syncthreads();
  }

  const int mh = (lane >> 4) << 3;
  const int cl = lane & 15;
  for (int mt = 0; mt < 4; ++mt)
    for (int nt = 0; nt < 2; ++nt)
      for (int i = 0; i < 8; ++i) {
        int r = bm + wm + mt * 16 + mh + i;
        if (r < M) D[(size_t)r * Nc + bn + wn + nt * 16 + cl] = acc[mt][nt][i];
      }
}

// ---------------------------------------------------------------------------
// Small kernels
// ---------------------------------------------------------------------------
__global__ void k_fill(float* p, float v, int n) {
  int i = blockIdx.x * blockDim.x + threadIdx.x;
  if (i < n) p[i] = v;
}

// f32 -> bf16 convert with fused epilogue: 0=plain, 1=*mul, 2=relu
__global__ void k_cvt(bf16_t* __restrict__ dst, const float* __restrict__ src,
                      const float* __restrict__ mul, int n, int amode) {
  int i = blockIdx.x * blockDim.x + threadIdx.x;
  if (i >= n) return;
  float f = src[i];
  if (amode == 1)      f *= mul[i];
  else if (amode == 2) f = fmaxf(f, 0.0f);
  dst[i] = to_bf16(f);
}

// W[K][Nc] f32 -> dst[Nc][K] bf16 (transpose so B tiles are TDM row-copyable)
__global__ void k_cvtT(bf16_t* __restrict__ dst, const float* __restrict__ W,
                       int K, int Nc) {
  int i = blockIdx.x * blockDim.x + threadIdx.x;
  if (i >= K * Nc) return;
  int n = i / K, k = i - n * K;
  dst[(size_t)n * K + k] = to_bf16(W[(size_t)k * Nc + n]);
}

__global__ void k_deg_edge(float* deg, const int* __restrict__ dst,
                           const float* __restrict__ w, int e) {
  int i = blockIdx.x * blockDim.x + threadIdx.x;
  if (i < e) atomicAdd(&deg[dst[i]], w[i]);
}

__global__ void k_dinv(float* d, int n) {
  int i = blockIdx.x * blockDim.x + threadIdx.x;
  if (i < n) d[i] = rsqrtf(fmaxf(d[i], 1e-12f));
}

__global__ void k_agg_self(float* agg, const float* __restrict__ x,
                           const float* __restrict__ dinv, int total) {
  int i = blockIdx.x * blockDim.x + threadIdx.x;
  if (i >= total) return;
  int n = i / (FEAT * PER);
  agg[i] = dinv[n] * dinv[n] * x[i];
}

__global__ void k_agg_edge(float* agg, const float* __restrict__ x,
                           const float* __restrict__ dinv,
                           const int* __restrict__ src, const int* __restrict__ dst,
                           const float* __restrict__ w, int total) {
  int i = blockIdx.x * blockDim.x + threadIdx.x;
  if (i >= total) return;
  int e = i / (FEAT * PER);
  int j = i - e * (FEAT * PER);
  int s = src[e], d = dst[e];
  float nm = dinv[s] * w[e] * dinv[d];
  atomicAdd(&agg[(size_t)d * (FEAT * PER) + j], nm * x[(size_t)s * (FEAT * PER) + j]);
}

__global__ void k_softmax(float* probs, const float* __restrict__ att) {
  if (threadIdx.x == 0 && blockIdx.x == 0) {
    float m = att[0];
    for (int i = 1; i < PER; ++i) m = fmaxf(m, att[i]);
    float e[PER]; float s = 0.f;
    for (int i = 0; i < PER; ++i) { e[i] = expf(att[i] - m); s += e[i]; }
    for (int i = 0; i < PER; ++i) probs[i] = e[i] / s;
  }
}

// M[f,c] = sum_k Wg[f,k]*Wl[k,c]; c[c] = sum_k bg[k]*Wl[k,c] + bl[c]  (top half)
__global__ void k_fold(const float* __restrict__ Wg, const float* __restrict__ bg,
                       const float* __restrict__ Wl, const float* __restrict__ bl,
                       float* Mo, float* co) {
  int t = blockIdx.x * blockDim.x + threadIdx.x;
  if (t >= 5 * CH) return;
  int f = t >> 9;
  int c = t & (CH - 1);
  const float* v = (f < FEAT) ? (Wg + (size_t)f * CH) : bg;
  float s = 0.f;
  for (int k = 0; k < CH; ++k) s = fmaf(v[k], Wl[(size_t)k * CH + c], s);
  if (f < FEAT) Mo[f * CH + c] = s;
  else          co[c] = s + bl[c];
}

__global__ void k_gate_zr(float* Z, float* R, const float* __restrict__ agg,
                          const float* __restrict__ Mz, const float* __restrict__ cz,
                          const float* __restrict__ Mr, const float* __restrict__ cr, int t) {
  int i = blockIdx.x * blockDim.x + threadIdx.x;
  if (i >= N_NODES * CH) return;
  int n = i >> 9, c = i & (CH - 1);
  const float* a = agg + (size_t)n * (FEAT * PER) + t;
  float a0 = a[0], a1 = a[PER], a2 = a[2 * PER], a3 = a[3 * PER];
  float dz = a0 * Mz[c] + a1 * Mz[CH + c] + a2 * Mz[2 * CH + c] + a3 * Mz[3 * CH + c] + cz[c];
  float dr = a0 * Mr[c] + a1 * Mr[CH + c] + a2 * Mr[2 * CH + c] + a3 * Mr[3 * CH + c] + cr[c];
  float zv = Z[i] + dz, rv = R[i] + dr;
  Z[i] = 1.0f / (1.0f + expf(-zv));
  R[i] = 1.0f / (1.0f + expf(-rv));
}

__global__ void k_update(float* H, float* acc, const float* __restrict__ HT,
                         const float* __restrict__ Z, const float* __restrict__ agg,
                         const float* __restrict__ Mh, const float* __restrict__ ch,
                         const float* __restrict__ probs, int t) {
  int i = blockIdx.x * blockDim.x + threadIdx.x;
  if (i >= N_NODES * CH) return;
  int n = i >> 9, c = i & (CH - 1);
  const float* a = agg + (size_t)n * (FEAT * PER) + t;
  float dh = a[0] * Mh[c] + a[PER] * Mh[CH + c] + a[2 * PER] * Mh[2 * CH + c]
           + a[3 * PER] * Mh[3 * CH + c] + ch[c];
  float ht = tanhf(HT[i] + dh);
  float z = Z[i], h = H[i];
  float hn = z * h + (1.0f - z) * ht;
  H[i] = hn;
  acc[i] += probs[t] * hn;
}

__global__ void k_bias_relu(float* hid, const float* __restrict__ b1) {
  int i = blockIdx.x * blockDim.x + threadIdx.x;
  if (i < N_NODES * HIDN) hid[i] = fmaxf(hid[i] + b1[i & (HIDN - 1)], 0.0f);
}

__global__ void k_out(float* __restrict__ out, const float* __restrict__ hid,
                      const float* __restrict__ W2, const float* __restrict__ b2) {
  int i = blockIdx.x * blockDim.x + threadIdx.x;
  if (i >= N_NODES * OUTD) return;
  int n = i / OUTD, o = i - n * OUTD;
  float s = b2[o];
  for (int h = 0; h < HIDN; ++h)
    s = fmaf(hid[(size_t)n * HIDN + h], W2[h * OUTD + o], s);
  out[i] = s;
}

// ---------------------------------------------------------------------------
extern "C" void kernel_launch(void* const* d_in, const int* in_sizes, int n_in,
                              void* d_out, int out_size, void* d_ws, size_t ws_size,
                              hipStream_t stream) {
  (void)in_sizes; (void)n_in; (void)out_size; (void)ws_size;

  const float* x    = (const float*)d_in[0];
  const int*   ei   = (const int*)d_in[1];
  const float* ew   = (const float*)d_in[2];
  const float* att  = (const float*)d_in[3];
  const float* Wz_g = (const float*)d_in[4];
  const float* bz_g = (const float*)d_in[5];
  const float* Wz_l = (const float*)d_in[6];
  const float* bz_l = (const float*)d_in[7];
  const float* Wr_g = (const float*)d_in[8];
  const float* br_g = (const float*)d_in[9];
  const float* Wr_l = (const float*)d_in[10];
  const float* br_l = (const float*)d_in[11];
  const float* Wh_g = (const float*)d_in[12];
  const float* bh_g = (const float*)d_in[13];
  const float* Wh_l = (const float*)d_in[14];
  const float* bh_l = (const float*)d_in[15];
  const float* W1   = (const float*)d_in[16];
  const float* b1   = (const float*)d_in[17];
  const float* W2   = (const float*)d_in[18];
  const float* b2   = (const float*)d_in[19];
  const int* esrc = ei;
  const int* edst = ei + N_EDGES;

  float* out  = (float*)d_out;
  float* hout = out;
  float* acc  = out + (size_t)N_NODES * OUTD;   // out_hidden lives here

  float* ws = (float*)d_ws;
  size_t off = 0;
  auto take = [&](size_t n) { float* p = ws + off; off += (n + 63) & ~(size_t)63; return p; };
  auto take_bf = [&](size_t n) { return (bf16_t*)take((n + 1) / 2); };

  float* dinv  = take(N_NODES);
  float* probs = take(PER);
  float* Mz = take(FEAT * CH); float* Mr = take(FEAT * CH); float* Mh = take(FEAT * CH);
  float* cz = take(CH);        float* cr = take(CH);        float* ch = take(CH);
  float* agg = take((size_t)N_NODES * FEAT * PER);
  float* H   = take((size_t)N_NODES * CH);
  float* Z   = take((size_t)N_NODES * CH);
  float* R   = take((size_t)N_NODES * CH);
  float* HT  = take((size_t)N_NODES * CH);        // also head hidden [N,HIDN]
  bf16_t* Ab  = take_bf((size_t)N_NODES * CH);    // bf16 A operand (H / H*r / relu(acc))
  bf16_t* Wzt = take_bf((size_t)CH * CH);         // [Nc][K] bf16
  bf16_t* Wrt = take_bf((size_t)CH * CH);
  bf16_t* Wht = take_bf((size_t)CH * CH);
  bf16_t* W1t = take_bf((size_t)HIDN * CH);

  const int TB = 256;
  auto cdiv = [](long long a, long long b) { return (int)((a + b - 1) / b); };

  // --- graph normalization + feature aggregation (once, over all T) ---
  k_fill<<<cdiv(N_NODES, TB), TB, 0, stream>>>(dinv, 1.0f, N_NODES);
  k_deg_edge<<<cdiv(N_EDGES, TB), TB, 0, stream>>>(dinv, edst, ew, N_EDGES);
  k_dinv<<<cdiv(N_NODES, TB), TB, 0, stream>>>(dinv, N_NODES);
  k_agg_self<<<cdiv((long long)N_NODES * FEAT * PER, TB), TB, 0, stream>>>(
      agg, x, dinv, N_NODES * FEAT * PER);
  k_agg_edge<<<cdiv((long long)N_EDGES * FEAT * PER, TB), TB, 0, stream>>>(
      agg, x, dinv, esrc, edst, ew, N_EDGES * FEAT * PER);
  k_softmax<<<1, 32, 0, stream>>>(probs, att);

  // --- fold GCN weights through top half of linear: M = Wg @ Wl_top [4,512] ---
  k_fold<<<cdiv(5 * CH, TB), TB, 0, stream>>>(Wz_g, bz_g, Wz_l, bz_l, Mz, cz);
  k_fold<<<cdiv(5 * CH, TB), TB, 0, stream>>>(Wr_g, br_g, Wr_l, br_l, Mr, cr);
  k_fold<<<cdiv(5 * CH, TB), TB, 0, stream>>>(Wh_g, bh_g, Wh_l, bh_l, Mh, ch);

  // --- pre-convert bottom-half weights (+W1) to transposed bf16 ---
  k_cvtT<<<cdiv((long long)CH * CH, TB), TB, 0, stream>>>(Wzt, Wz_l + (size_t)CH * CH, CH, CH);
  k_cvtT<<<cdiv((long long)CH * CH, TB), TB, 0, stream>>>(Wrt, Wr_l + (size_t)CH * CH, CH, CH);
  k_cvtT<<<cdiv((long long)CH * CH, TB), TB, 0, stream>>>(Wht, Wh_l + (size_t)CH * CH, CH, CH);
  k_cvtT<<<cdiv((long long)CH * HIDN, TB), TB, 0, stream>>>(W1t, W1, CH, HIDN);

  k_fill<<<cdiv((long long)N_NODES * CH, TB), TB, 0, stream>>>(H, 0.0f, N_NODES * CH);
  k_fill<<<cdiv((long long)N_NODES * CH, TB), TB, 0, stream>>>(acc, 0.0f, N_NODES * CH);

  // --- recurrent loop ---
  dim3 gg(CH / BN, cdiv(N_NODES, BM));
  const int NCgrid = cdiv((long long)N_NODES * CH, TB);
  for (int t = 0; t < PER; ++t) {
    k_cvt<<<NCgrid, TB, 0, stream>>>(Ab, H, nullptr, N_NODES * CH, 0);
    k_gemm_wmma<<<gg, 256, 0, stream>>>(Ab, Wzt, Z, N_NODES, CH, CH);
    k_gemm_wmma<<<gg, 256, 0, stream>>>(Ab, Wrt, R, N_NODES, CH, CH);
    k_gate_zr<<<NCgrid, TB, 0, stream>>>(Z, R, agg, Mz, cz, Mr, cr, t);
    k_cvt<<<NCgrid, TB, 0, stream>>>(Ab, H, R, N_NODES * CH, 1);       // A = H*r
    k_gemm_wmma<<<gg, 256, 0, stream>>>(Ab, Wht, HT, N_NODES, CH, CH);
    k_update<<<NCgrid, TB, 0, stream>>>(H, acc, HT, Z, agg, Mh, ch, probs, t);
  }

  // --- head: relu(acc) @ W1 (+b1, relu) @ W2 + b2 ---
  dim3 hg(HIDN / BN, cdiv(N_NODES, BM));
  k_cvt<<<NCgrid, TB, 0, stream>>>(Ab, acc, nullptr, N_NODES * CH, 2);
  k_gemm_wmma<<<hg, 256, 0, stream>>>(Ab, W1t, HT, N_NODES, CH, HIDN);
  k_bias_relu<<<cdiv((long long)N_NODES * HIDN, TB), TB, 0, stream>>>(HT, b1);
  k_out<<<cdiv((long long)N_NODES * OUTD, TB), TB, 0, stream>>>(hout, HT, W2, b2);
}
